// LanguageWithAttention_35880156791547
// MI455X (gfx1250) — compile-verified, matrix operand order checked by reading, not run on hardware
//
#include <hip/hip_runtime.h>
#include <hip/hip_bf16.h>
#include <stddef.h>

// ---------------------------------------------------------------------------
// CDNA5 (gfx1250) WMMA helpers: v_wmma_f32_16x16x32_bf16
// A: 16x32 (MxK)  B: 32x16 (KxN)  C/D: 16x16 f32
// All operands are staged as pre-packed bf16-pair dwords so fragment assembly
// is pure 128-bit loads (2x b128 per fragment), no per-use conversion VALU.
// ---------------------------------------------------------------------------
typedef __attribute__((ext_vector_type(16))) __bf16    v16bf;
typedef __attribute__((ext_vector_type(8)))  float     v8f;
typedef __attribute__((ext_vector_type(4)))  unsigned  v4u;

union FragU {
    v16bf    bf;
    v4u      u4[2];
    unsigned u[8];
};

__device__ __forceinline__ unsigned f2bf1(float f) {
    union { float f; unsigned u; } v; v.f = f;
    unsigned r = v.u + 0x7FFFu + ((v.u >> 16) & 1u);   // round-to-nearest-even
    return r >> 16;
}
__device__ __forceinline__ unsigned pack2bf(float lo, float hi) {
    return (f2bf1(lo) & 0xFFFFu) | (f2bf1(hi) << 16);
}

// A fragment from a bf16-pair-packed row-major matrix (LDS).
// stride2 = dwords per row (multiple of 4). Layout per ISA 16-bit A 16x32:
// lane&15 = row M; half-wave selects K {0-7,16-23} vs {8-15,24-31}.
__device__ __forceinline__ FragU load_a_packed(const unsigned* Ap, int stride2, int kb, int lane) {
    const unsigned* rp = Ap + (lane & 15) * stride2 + (kb >> 1) + ((lane >> 4) << 2);
    FragU a;
    a.u4[0] = *(const v4u*)(rp);
    a.u4[1] = *(const v4u*)(rp + 8);
    return a;
}

// B fragment from tiled packed weights: Bt[kt][n][k2l], k2l = 0..15.
// col must already include (lane & 15).
__device__ __forceinline__ FragU load_b_tiled(const unsigned* Bt, int N, int kt, int col, int lane) {
    const unsigned* base = Bt + (((size_t)kt * N + col) << 4) + ((lane >> 4) << 2);
    FragU b;
    b.u4[0] = *(const v4u*)(base);
    b.u4[1] = *(const v4u*)(base + 8);
    return b;
}

__device__ __forceinline__ v8f wmma_bf(const FragU& a, const FragU& b, v8f c) {
    return __builtin_amdgcn_wmma_f32_16x16x32_bf16(false, a.bf, false, b.bf,
                                                   (short)0, c, false, false);
}

__device__ __forceinline__ float sigf(float x) { return 1.0f / (1.0f + __expf(-x)); }

// ---------------------------------------------------------------------------
// Weight packing: W (N,K) fp32 -> Bt (KT, N, 16) u32 bf16-pairs (K zero-padded)
// ---------------------------------------------------------------------------
__global__ void pack_w_tiled(const float* __restrict__ W, unsigned* __restrict__ Bt,
                             int N, int K, int KT) {
    int total = KT * N * 16;
    for (int i = blockIdx.x * blockDim.x + threadIdx.x; i < total;
         i += gridDim.x * blockDim.x) {
        int kt  = i / (N * 16);
        int rem = i - kt * N * 16;
        int n   = rem >> 4;
        int k2l = rem & 15;
        int k   = (kt * 16 + k2l) * 2;
        float f0 = (k     < K) ? W[(size_t)n * K + k]     : 0.0f;
        float f1 = (k + 1 < K) ? W[(size_t)n * K + k + 1] : 0.0f;
        Bt[i] = pack2bf(f0, f1);
    }
}

// Concatenated LSTM weight: logical K: [0,428)=W_ih, [428,460)=W_hh, pad->480 (KT=15)
__global__ void pack_ihh_tiled(const float* __restrict__ Wih, const float* __restrict__ Whh,
                               unsigned* __restrict__ Bt) {
    const int total = 15 * 128 * 16;
    for (int i = blockIdx.x * blockDim.x + threadIdx.x; i < total;
         i += gridDim.x * blockDim.x) {
        int kt  = i >> 11;          // 128*16 = 2048
        int rem = i & 2047;
        int n   = rem >> 4;
        int k2l = rem & 15;
        float f[2];
#pragma unroll
        for (int j = 0; j < 2; ++j) {
            int k = (kt * 16 + k2l) * 2 + j;
            f[j] = (k < 428) ? Wih[(size_t)n * 428 + k]
                 : (k < 460) ? Whh[(size_t)n * 32 + (k - 428)]
                 : 0.0f;
        }
        Bt[i] = pack2bf(f[0], f[1]);
    }
}

// ---------------------------------------------------------------------------
// Encoder linear: 16 rows x 128 cols per block, K=300 padded to 320 (KT=10).
// mode 0: grid (adds relu(onehot linear)), mode 1: inventory, mode 2: goal.
// ---------------------------------------------------------------------------
__global__ void enc_lin_kernel(const float* __restrict__ src, const unsigned* __restrict__ Wt,
                               const float* __restrict__ bias,
                               const float* __restrict__ oh, const float* __restrict__ W_oh,
                               const float* __restrict__ b_oh,
                               float* __restrict__ enc, int mode) {
    __shared__ __align__(16) unsigned Asp[16][160];
    const int r0 = blockIdx.x * 16;
    const int tid = threadIdx.x;
    for (int i = tid; i < 16 * 160; i += 256) {
        int row = i / 160, k2 = i - row * 160;
        unsigned pv = 0u;
        if (k2 < 150) {
            const float* sp = src + (size_t)(r0 + row) * 300 + k2 * 2;
            pv = pack2bf(sp[0], sp[1]);
        }
        Asp[row][k2] = pv;
    }
    __syncthreads();
    const int wid = tid >> 5, lane = tid & 31;
    const int col = wid * 16 + (lane & 15);
    v8f acc = {};
#pragma unroll
    for (int kt = 0; kt < 10; ++kt) {
        FragU a = load_a_packed(&Asp[0][0], 160, kt * 32, lane);
        FragU b = load_b_tiled(Wt, 128, kt, col, lane);
        acc = wmma_bf(a, b, acc);
    }
    const int rbase = (lane >> 4) * 8;
    const float bc = bias[col];
#pragma unroll
    for (int r = 0; r < 8; ++r) {
        int rr = r0 + rbase + r;
        float v = fmaxf(acc[r] + bc, 0.0f);
        int bb, pp;
        if (mode == 0)      { bb = rr / 25; pp = rr % 25; }
        else if (mode == 1) { bb = rr / 10; pp = 25 + rr % 10; }
        else                { bb = rr;      pp = 35; }
        if (mode == 0) {
            float c2 = b_oh[col];
#pragma unroll
            for (int s = 0; s < 7; ++s)
                c2 += oh[(size_t)rr * 7 + s] * W_oh[col * 7 + s];
            v += fmaxf(c2, 0.0f);
        }
        enc[((size_t)bb * 36 + pp) * 128 + col] = v;
    }
}

// att1 = enc @ W_enc_att^T + b : M = B*36, K=128 (KT=4), N=128
__global__ void att1_kernel(const float* __restrict__ enc, const unsigned* __restrict__ Wt,
                            const float* __restrict__ bias, float* __restrict__ att1) {
    __shared__ __align__(16) unsigned Asp[16][64];
    const int r0 = blockIdx.x * 16;
    const int tid = threadIdx.x;
    for (int i = tid; i < 16 * 64; i += 256) {
        int row = i >> 6, k2 = i & 63;
        const float* sp = enc + (size_t)(r0 + row) * 128 + k2 * 2;
        Asp[row][k2] = pack2bf(sp[0], sp[1]);
    }
    __syncthreads();
    const int wid = tid >> 5, lane = tid & 31;
    const int col = wid * 16 + (lane & 15);
    v8f acc = {};
#pragma unroll
    for (int kt = 0; kt < 4; ++kt) {
        FragU a = load_a_packed(&Asp[0][0], 64, kt * 32, lane);
        FragU b = load_b_tiled(Wt, 128, kt, col, lane);
        acc = wmma_bf(a, b, acc);
    }
    const int rbase = (lane >> 4) * 8;
    const float bc = bias[col];
#pragma unroll
    for (int r = 0; r < 8; ++r)
        att1[(size_t)(r0 + rbase + r) * 128 + col] = acc[r] + bc;
}

// ---------------------------------------------------------------------------
// Decoder: 16 batch rows per block, all T steps, state in LDS.
// Keeps fp32 h/c plus a packed-bf16 shadow (hp_s) for WMMA A-operands.
// Emits packed h_t to h_allp; alphas + final h to d_out.
// ---------------------------------------------------------------------------
__global__ void decoder_kernel(const float* __restrict__ enc, const float* __restrict__ att1,
                               const int* __restrict__ caps, int L, int T,
                               const float* __restrict__ emb_table,
                               const unsigned* __restrict__ Wt_dec_att, const float* __restrict__ b_dec_att,
                               const float* __restrict__ W_full, const float* __restrict__ b_full,
                               const unsigned* __restrict__ Wt_fbeta, const float* __restrict__ b_fbeta,
                               const unsigned* __restrict__ Wt_ihh, const float* __restrict__ b_ih,
                               const float* __restrict__ b_hh,
                               const float* __restrict__ W_init_h, const float* __restrict__ b_init_h,
                               const float* __restrict__ W_init_c, const float* __restrict__ b_init_c,
                               unsigned* __restrict__ h_allp, float* __restrict__ alphas_out,
                               float* __restrict__ h_out) {
    __shared__ float h_s[16][32];
    __shared__ float c_s[16][32];
    __shared__ __align__(16) unsigned hp_s[16][16];    // packed bf16 h
    __shared__ float att2_s[16][128];                  // also mean_enc in prologue
    __shared__ float ea_s[16][36];                     // e, then alpha
    __shared__ float ag_s[16][128];                    // awe, then gate*awe
    __shared__ __align__(16) unsigned xp_s[16][240];   // packed [emb|gate*awe|h|0]
    __shared__ float g_s[16][128];                     // LSTM gates

    const int b0 = blockIdx.x * 16;
    const int tid = threadIdx.x;
    const int wid = tid >> 5, lane = tid & 31;
    const int ntile = wid * 16;
    const int colw  = ntile + (lane & 15);
    const int rbase = (lane >> 4) * 8;

    // mean_enc -> att2_s
    for (int i = tid; i < 16 * 128; i += 256) {
        int row = i >> 7, n = i & 127;
        const float* ep = enc + ((size_t)(b0 + row) * 36) * 128 + n;
        float s = 0.0f;
#pragma unroll 4
        for (int p = 0; p < 36; ++p) s += ep[p * 128];
        att2_s[row][n] = s * (1.0f / 36.0f);
    }
    __syncthreads();
    // h0 / c0
    for (int i = tid; i < 16 * 32; i += 256) {
        int row = i >> 5, k = i & 31;
        float sh = b_init_h[k], sc = b_init_c[k];
        for (int n = 0; n < 128; ++n) {
            float m = att2_s[row][n];
            sh += m * W_init_h[k * 128 + n];
            sc += m * W_init_c[k * 128 + n];
        }
        h_s[row][k] = sh;
        c_s[row][k] = sc;
    }
    __syncthreads();
    if (tid < 256) {
        int row = tid >> 4, k2 = tid & 15;
        hp_s[row][k2] = pack2bf(h_s[row][2 * k2], h_s[row][2 * k2 + 1]);
    }
    __syncthreads();

    for (int t = 0; t < T; ++t) {
        // shared A fragment of h for att2 + gate WMMAs (K=32, single tile)
        FragU ah = load_a_packed(&hp_s[0][0], 16, 0, lane);
        // (a) att2 = h @ W_dec_att^T + b
        {
            FragU b = load_b_tiled(Wt_dec_att, 128, 0, colw, lane);
            v8f acc = {};
            acc = wmma_bf(ah, b, acc);
            float bb = b_dec_att[colw];
#pragma unroll
            for (int r = 0; r < 8; ++r)
                att2_s[rbase + r][colw] = acc[r] + bb;
        }
        __syncthreads();
        // (b) e[row][p] = relu(att1 + att2) . W_full + b_full
        for (int i = tid; i < 16 * 36; i += 256) {
            int row = i / 36, p = i - row * 36;
            const float* a1 = att1 + ((size_t)(b0 + row) * 36 + p) * 128;
            float s = b_full[0];
#pragma unroll 4
            for (int n = 0; n < 128; ++n)
                s += fmaxf(a1[n] + att2_s[row][n], 0.0f) * W_full[n];
            ea_s[row][p] = s;
        }
        __syncthreads();
        // softmax over 36 (one thread per row)
        if (tid < 16) {
            int row = tid;
            float m = -1e30f;
            for (int p = 0; p < 36; ++p) m = fmaxf(m, ea_s[row][p]);
            float s = 0.0f;
            for (int p = 0; p < 36; ++p) {
                float ev = __expf(ea_s[row][p] - m);
                ea_s[row][p] = ev; s += ev;
            }
            float inv = 1.0f / s;
            float* ao = alphas_out + ((size_t)(b0 + row) * T + t) * 36;
            for (int p = 0; p < 36; ++p) {
                float al = ea_s[row][p] * inv;
                ea_s[row][p] = al;
                ao[p] = al;
            }
        }
        __syncthreads();
        // (c) awe
        for (int i = tid; i < 16 * 128; i += 256) {
            int row = i >> 7, n = i & 127;
            const float* ep = enc + ((size_t)(b0 + row) * 36) * 128 + n;
            float s = 0.0f;
#pragma unroll 4
            for (int p = 0; p < 36; ++p) s += ea_s[row][p] * ep[p * 128];
            ag_s[row][n] = s;
        }
        __syncthreads();
        // gate = sigmoid(h @ W_fbeta^T + b); ag *= gate
        {
            FragU b = load_b_tiled(Wt_fbeta, 128, 0, colw, lane);
            v8f acc = {};
            acc = wmma_bf(ah, b, acc);
            float bb = b_fbeta[colw];
#pragma unroll
            for (int r = 0; r < 8; ++r)
                ag_s[rbase + r][colw] *= sigf(acc[r] + bb);
        }
        __syncthreads();
        // (d) build packed x = [emb | gate*awe | h | 0]; pairs never straddle regions
        for (int i = tid; i < 16 * 240; i += 256) {
            int row = i / 240, c2 = i - row * 240;
            unsigned pv;
            if (c2 < 150) {
                int tok = caps[(size_t)(b0 + row) * L + t];
                const float* ep = emb_table + (size_t)tok * 300 + c2 * 2;
                pv = pack2bf(ep[0], ep[1]);
            } else if (c2 < 214) {
                int j = (c2 - 150) * 2;
                pv = pack2bf(ag_s[row][j], ag_s[row][j + 1]);
            } else if (c2 < 230) {
                pv = hp_s[row][c2 - 214];
            } else {
                pv = 0u;
            }
            xp_s[row][c2] = pv;
        }
        __syncthreads();
        // (e) gates = x @ [W_ih;W_hh]^T + b_ih + b_hh  (15 WMMAs per wave)
        {
            v8f acc = {};
#pragma unroll
            for (int kt = 0; kt < 15; ++kt) {
                FragU a = load_a_packed(&xp_s[0][0], 240, kt * 32, lane);
                FragU b = load_b_tiled(Wt_ihh, 128, kt, colw, lane);
                acc = wmma_bf(a, b, acc);
            }
            float bb = b_ih[colw] + b_hh[colw];
#pragma unroll
            for (int r = 0; r < 8; ++r)
                g_s[rbase + r][colw] = acc[r] + bb;
        }
        __syncthreads();
        // (f) LSTM elementwise
        for (int i = tid; i < 16 * 32; i += 256) {
            int row = i >> 5, k = i & 31;
            float ig = g_s[row][k], fg = g_s[row][32 + k];
            float gg = g_s[row][64 + k], og = g_s[row][96 + k];
            float cn = sigf(fg) * c_s[row][k] + sigf(ig) * tanhf(gg);
            float hn = sigf(og) * tanhf(cn);
            c_s[row][k] = cn;
            h_s[row][k] = hn;
        }
        __syncthreads();
        // repack h + emit packed h_t for the bulk vocab GEMM
        if (tid < 256) {
            int row = tid >> 4, k2 = tid & 15;
            unsigned pv = pack2bf(h_s[row][2 * k2], h_s[row][2 * k2 + 1]);
            hp_s[row][k2] = pv;
            h_allp[((size_t)(b0 + row) * T + t) * 16 + k2] = pv;
        }
        __syncthreads();
    }
    // final h (fp32)
    for (int i = tid; i < 16 * 32; i += 256) {
        int row = i >> 5, k = i & 31;
        h_out[(size_t)(b0 + row) * 32 + k] = h_s[row][k];
    }
}

// ---------------------------------------------------------------------------
// Vocab projection: M = B*T rows, K=32 (one WMMA per tile), N=2000.
// A comes pre-packed from the decoder; B pre-packed tiled -> pure b128 loads.
// ---------------------------------------------------------------------------
__global__ void preds_kernel(const unsigned* __restrict__ h_allp, const unsigned* __restrict__ Wt_fc,
                             const float* __restrict__ b_fc, float* __restrict__ preds, int Mtot) {
    __shared__ __align__(16) unsigned Asp[16][16];
    const int r0 = blockIdx.x * 16;
    const int tid = threadIdx.x;
    if (tid < 256) {
        int row = tid >> 4, k2 = tid & 15;
        int rr = r0 + row;
        if (rr >= Mtot) rr = Mtot - 1;
        Asp[row][k2] = h_allp[(size_t)rr * 16 + k2];
    }
    __syncthreads();
    const int wid = tid >> 5, lane = tid & 31;
    FragU a = load_a_packed(&Asp[0][0], 16, 0, lane);
    const int rbase = (lane >> 4) * 8, c0 = lane & 15;
    for (int vt = wid; vt < 125; vt += 8) {
        int col = vt * 16 + c0;
        FragU b = load_b_tiled(Wt_fc, 2000, 0, col, lane);
        v8f acc = {};
        acc = wmma_bf(a, b, acc);
        float bc = b_fc[col];
#pragma unroll
        for (int r = 0; r < 8; ++r) {
            int rr = r0 + rbase + r;
            if (rr < Mtot) preds[(size_t)rr * 2000 + col] = acc[r] + bc;
        }
    }
}

// ---------------------------------------------------------------------------
// Host launcher
// ---------------------------------------------------------------------------
extern "C" void kernel_launch(void* const* d_in, const int* in_sizes, int n_in,
                              void* d_out, int out_size, void* d_ws, size_t ws_size,
                              hipStream_t stream) {
    const float* grid_emb  = (const float*)d_in[0];
    const float* grid_oh   = (const float*)d_in[1];
    const float* inv       = (const float*)d_in[2];
    const float* goal      = (const float*)d_in[3];
    const int*   caps      = (const int*)d_in[4];
    // d_in[5] = caption_length (device scalar; T derived from out_size instead)
    const float* W_embed   = (const float*)d_in[6];
    const float* b_embed   = (const float*)d_in[7];
    const float* W_onehot  = (const float*)d_in[8];
    const float* b_onehot  = (const float*)d_in[9];
    const float* W_inv     = (const float*)d_in[10];
    const float* b_inv     = (const float*)d_in[11];
    const float* W_goal    = (const float*)d_in[12];
    const float* b_goal    = (const float*)d_in[13];
    const float* W_enc_att = (const float*)d_in[14];
    const float* b_enc_att = (const float*)d_in[15];
    const float* W_dec_att = (const float*)d_in[16];
    const float* b_dec_att = (const float*)d_in[17];
    const float* W_full    = (const float*)d_in[18];
    const float* b_full    = (const float*)d_in[19];
    const float* emb_table = (const float*)d_in[20];
    const float* W_ih      = (const float*)d_in[21];
    const float* b_ih      = (const float*)d_in[22];
    const float* W_hh      = (const float*)d_in[23];
    const float* b_hh      = (const float*)d_in[24];
    const float* W_init_h  = (const float*)d_in[25];
    const float* b_init_h  = (const float*)d_in[26];
    const float* W_init_c  = (const float*)d_in[27];
    const float* b_init_c  = (const float*)d_in[28];
    const float* W_fbeta   = (const float*)d_in[29];
    const float* b_fbeta   = (const float*)d_in[30];
    const float* W_fc      = (const float*)d_in[31];
    const float* b_fc      = (const float*)d_in[32];

    const int B = in_sizes[3] / 300;                 // goal is (B, 300)
    const int L = in_sizes[4] / B;                   // captions (B, L)
    const int T = (int)(((size_t)out_size - (size_t)32 * B) / ((size_t)B * 2036));

    // workspace carve-up (all chunks 16B-aligned)
    char* w = (char*)d_ws;
    float*    enc    = (float*)w;    w += (size_t)B * 36 * 128 * 4;
    float*    att1   = (float*)w;    w += (size_t)B * 36 * 128 * 4;
    unsigned* h_allp = (unsigned*)w; w += (size_t)B * T * 16 * 4;
    unsigned* Wt_embed   = (unsigned*)w; w += 10 * 128 * 16 * 4;
    unsigned* Wt_inv     = (unsigned*)w; w += 10 * 128 * 16 * 4;
    unsigned* Wt_goal    = (unsigned*)w; w += 10 * 128 * 16 * 4;
    unsigned* Wt_enc_att = (unsigned*)w; w += 4 * 128 * 16 * 4;
    unsigned* Wt_dec_att = (unsigned*)w; w += 1 * 128 * 16 * 4;
    unsigned* Wt_fbeta   = (unsigned*)w; w += 1 * 128 * 16 * 4;
    unsigned* Wt_ihh     = (unsigned*)w; w += 15 * 128 * 16 * 4;
    unsigned* Wt_fc      = (unsigned*)w; w += 1 * 2000 * 16 * 4;
    (void)ws_size; (void)n_in;

    // pack weights (bf16-pair, (kt, n, k2l) tiles)
    pack_w_tiled<<<80, 256, 0, stream>>>(W_embed,   Wt_embed,   128, 300, 10);
    pack_w_tiled<<<80, 256, 0, stream>>>(W_inv,     Wt_inv,     128, 300, 10);
    pack_w_tiled<<<80, 256, 0, stream>>>(W_goal,    Wt_goal,    128, 300, 10);
    pack_w_tiled<<<32, 256, 0, stream>>>(W_enc_att, Wt_enc_att, 128, 128, 4);
    pack_w_tiled<<<8,  256, 0, stream>>>(W_dec_att, Wt_dec_att, 128, 32,  1);
    pack_w_tiled<<<8,  256, 0, stream>>>(W_fbeta,   Wt_fbeta,   128, 32,  1);
    pack_ihh_tiled<<<120, 256, 0, stream>>>(W_ih, W_hh, Wt_ihh);
    pack_w_tiled<<<125, 256, 0, stream>>>(W_fc, Wt_fc, 2000, 32, 1);

    // encoder
    enc_lin_kernel<<<B * 25 / 16, 256, 0, stream>>>(grid_emb, Wt_embed, b_embed,
                                                    grid_oh, W_onehot, b_onehot, enc, 0);
    enc_lin_kernel<<<B * 10 / 16, 256, 0, stream>>>(inv, Wt_inv, b_inv,
                                                    nullptr, nullptr, nullptr, enc, 1);
    enc_lin_kernel<<<B / 16, 256, 0, stream>>>(goal, Wt_goal, b_goal,
                                               nullptr, nullptr, nullptr, enc, 2);
    att1_kernel<<<B * 36 / 16, 256, 0, stream>>>(enc, Wt_enc_att, b_enc_att, att1);

    float* preds_out  = (float*)d_out;
    float* alphas_out = preds_out + (size_t)B * T * 2000;
    float* h_out      = alphas_out + (size_t)B * T * 36;

    // sequential recurrence (batch-parallel, no inter-block sync needed)
    decoder_kernel<<<B / 16, 256, 0, stream>>>(enc, att1, caps, L, T, emb_table,
                                               Wt_dec_att, b_dec_att, W_full, b_full,
                                               Wt_fbeta, b_fbeta, Wt_ihh, b_ih, b_hh,
                                               W_init_h, b_init_h, W_init_c, b_init_c,
                                               h_allp, alphas_out, h_out);

    // bulk vocab projection (fully parallel over B*T)
    const int Mtot = B * T;
    preds_kernel<<<(Mtot + 15) / 16, 256, 0, stream>>>(h_allp, Wt_fc, b_fc, preds_out, Mtot);
}
